// StackedRecurrentAttention_7524782702772
// MI455X (gfx1250) — compile-verified, matrix operand order checked by reading, not run on hardware
//
#include <hip/hip_runtime.h>
#include <math.h>

// ---------------- problem constants ----------------
#define T_STEPS 256
#define BATCH   64
#define HID     512
#define DCTX    512
#define SEQ     512
#define KIN     1024      // HID + DCTX
#define GATES   2048      // 4*HID
#define NLAYERS 2
#define NKT_IH  (KIN / 32)   // 32 k-tiles over cat
#define NKT_HH  (HID / 32)   // 16 k-tiles over h
#define NKT_ALL (NKT_IH + NKT_HH)

// ---------------- CDNA5 types ----------------
typedef __attribute__((ext_vector_type(16))) __bf16    v16bf;
typedef __attribute__((ext_vector_type(8)))  float     v8f;
typedef __attribute__((ext_vector_type(4)))  unsigned  uint4v;
typedef __attribute__((ext_vector_type(4)))  unsigned  u32x4;
typedef __attribute__((ext_vector_type(8)))  int       i32x8;
typedef __attribute__((ext_vector_type(4)))  int       i32x4;

union TileBF { v16bf v; uint4v u[2]; };

__device__ __forceinline__ unsigned short f2bf(float f) {
  unsigned u = __float_as_uint(f);
  u += 0x7FFFu + ((u >> 16) & 1u);          // round-to-nearest-even
  return (unsigned short)(u >> 16);
}
__device__ __forceinline__ float bflo(unsigned u) { return __uint_as_float(u << 16); }
__device__ __forceinline__ float bfhi(unsigned u) { return __uint_as_float(u & 0xFFFF0000u); }

__device__ __forceinline__ v8f wmma_bf16(v16bf a, v16bf b, v8f c) {
  return __builtin_amdgcn_wmma_f32_16x16x32_bf16(false, a, false, b, (short)0, c, false, false);
}
__device__ __forceinline__ float sigf(float x) { return 1.0f / (1.0f + __expf(-x)); }

// ==================================================================
// One-time (per call) conversion / packing kernels
// ==================================================================
__global__ void ctx2bf_kernel(const float* __restrict__ src,
                              unsigned short* __restrict__ dst, int n2) {
  int i = blockIdx.x * blockDim.x + threadIdx.x;
  if (i >= n2) return;
  dst[2 * i]     = f2bf(src[2 * i]);
  dst[2 * i + 1] = f2bf(src[2 * i + 1]);
}

// Pack W_ih [L][GATES][KIN] into WMMA-B lane layout:
// P[li][nt][kt][lane][j], nt = hgroup*4 + gate (gate-interleaved n remap)
__global__ void pack_wih_kernel(const float* __restrict__ W,
                                unsigned short* __restrict__ P) {
  unsigned gid = blockIdx.x * blockDim.x + threadIdx.x;
  const unsigned per_layer = (unsigned)GATES * KIN;
  if (gid >= (unsigned)NLAYERS * per_layer) return;
  unsigned li = gid / per_layer, r = gid % per_layer;
  unsigned j    = r & 15;
  unsigned lane = (r >> 4) & 31;
  unsigned tk   = r >> 9;
  unsigned kt   = tk & 31;
  unsigned nt   = tk >> 5;
  unsigned hi   = lane >> 4;
  unsigned k    = kt * 32 + ((j < 8) ? (j + 8 * hi) : (16 + (j - 8) + 8 * hi));
  unsigned gate = nt & 3, hg = nt >> 2;
  unsigned n    = gate * HID + hg * 16 + (lane & 15);
  P[gid] = f2bf(W[((size_t)li * GATES + n) * KIN + k]);
}

__global__ void pack_whh_kernel(const float* __restrict__ W,
                                unsigned short* __restrict__ P) {
  unsigned gid = blockIdx.x * blockDim.x + threadIdx.x;
  const unsigned per_layer = (unsigned)GATES * HID;
  if (gid >= (unsigned)NLAYERS * per_layer) return;
  unsigned li = gid / per_layer, r = gid % per_layer;
  unsigned j    = r & 15;
  unsigned lane = (r >> 4) & 31;
  unsigned tk   = r >> 9;
  unsigned kt   = tk & 15;
  unsigned nt   = tk >> 4;
  unsigned hi   = lane >> 4;
  unsigned k    = kt * 32 + ((j < 8) ? (j + 8 * hi) : (16 + (j - 8) + 8 * hi));
  unsigned gate = nt & 3, hg = nt >> 2;
  unsigned n    = gate * HID + hg * 16 + (lane & 15);
  P[gid] = f2bf(W[((size_t)li * GATES + n) * HID + k]);
}

__global__ void bias_sum_kernel(const float* __restrict__ bih,
                                const float* __restrict__ bhh,
                                float* __restrict__ bs, int n) {
  int i = blockIdx.x * blockDim.x + threadIdx.x;
  if (i < n) bs[i] = bih[i] + bhh[i];
}

// ==================================================================
// Attention: one block per batch element. scores -> softmax -> wctx.
// ==================================================================
__global__ __launch_bounds__(256)
void attn_kernel(const float* __restrict__ query,          // [B][HID]
                 const unsigned short* __restrict__ ctxb,  // [B][SEQ][DCTX] bf16
                 unsigned short* __restrict__ cat_bf,      // [B][KIN] bf16
                 float* __restrict__ attn_out,             // [B][SEQ] or null
                 float* __restrict__ wctx_out) {           // [B][DCTX] or null
  __shared__ float q[DCTX];
  __shared__ float p[SEQ];
  __shared__ float red[256];

  const int b    = blockIdx.x;
  const int tid  = threadIdx.x;
  const int lane = tid & 31;
  const int wv   = tid >> 5;

  float q0 = query[b * HID + 2 * tid];
  float q1 = query[b * HID + 2 * tid + 1];
  q[2 * tid] = q0; q[2 * tid + 1] = q1;
  cat_bf[(size_t)b * KIN + 2 * tid]     = f2bf(q0);
  cat_bf[(size_t)b * KIN + 2 * tid + 1] = f2bf(q1);
  __syncthreads();

  const unsigned short* cb = ctxb + (size_t)b * SEQ * DCTX;

  for (int s = wv; s < SEQ; s += 8) {
    const uint4v* pr = (const uint4v*)(cb + (size_t)s * DCTX + lane * 16);
    uint4v u0 = pr[0], u1 = pr[1];
    const float* qq = &q[lane * 16];
    float acc = 0.f;
    acc = fmaf(bflo(u0.x), qq[0],  acc); acc = fmaf(bfhi(u0.x), qq[1],  acc);
    acc = fmaf(bflo(u0.y), qq[2],  acc); acc = fmaf(bfhi(u0.y), qq[3],  acc);
    acc = fmaf(bflo(u0.z), qq[4],  acc); acc = fmaf(bfhi(u0.z), qq[5],  acc);
    acc = fmaf(bflo(u0.w), qq[6],  acc); acc = fmaf(bfhi(u0.w), qq[7],  acc);
    acc = fmaf(bflo(u1.x), qq[8],  acc); acc = fmaf(bfhi(u1.x), qq[9],  acc);
    acc = fmaf(bflo(u1.y), qq[10], acc); acc = fmaf(bfhi(u1.y), qq[11], acc);
    acc = fmaf(bflo(u1.z), qq[12], acc); acc = fmaf(bfhi(u1.z), qq[13], acc);
    acc = fmaf(bflo(u1.w), qq[14], acc); acc = fmaf(bfhi(u1.w), qq[15], acc);
    #pragma unroll
    for (int off = 16; off; off >>= 1) acc += __shfl_xor(acc, off, 32);
    if (lane == 0) p[s] = acc;
  }
  __syncthreads();

  float m = fmaxf(p[tid], p[tid + 256]);
  red[tid] = m; __syncthreads();
  for (int st = 128; st; st >>= 1) {
    if (tid < st) red[tid] = fmaxf(red[tid], red[tid + st]);
    __syncthreads();
  }
  m = red[0]; __syncthreads();
  float e0 = __expf(p[tid] - m), e1 = __expf(p[tid + 256] - m);
  p[tid] = e0; p[tid + 256] = e1;
  red[tid] = e0 + e1; __syncthreads();
  for (int st = 128; st; st >>= 1) {
    if (tid < st) red[tid] += red[tid + st];
    __syncthreads();
  }
  const float rinv = 1.f / red[0];
  __syncthreads();

  if (attn_out) {
    attn_out[(size_t)b * SEQ + tid]       = p[tid] * rinv;
    attn_out[(size_t)b * SEQ + tid + 256] = p[tid + 256] * rinv;
  }

  const int d0 = 2 * tid;
  float a0 = 0.f, a1 = 0.f;
  for (int s = 0; s < SEQ; ++s) {
    float ps = p[s];
    unsigned u = *(const unsigned*)(cb + (size_t)s * DCTX + d0);
    a0 = fmaf(ps, bflo(u), a0);
    a1 = fmaf(ps, bfhi(u), a1);
  }
  a0 *= rinv; a1 *= rinv;
  if (wctx_out) {
    wctx_out[(size_t)b * DCTX + d0]     = a0;
    wctx_out[(size_t)b * DCTX + d0 + 1] = a1;
  }
  cat_bf[(size_t)b * KIN + HID + d0]     = f2bf(a0);
  cat_bf[(size_t)b * KIN + HID + d0 + 1] = f2bf(a1);
}

// ==================================================================
// LSTM cell with TDM-staged weights (double-buffered LDS stages),
// WMMA bf16 GEMM, in-register gate activations.
// ==================================================================
__global__ __launch_bounds__(128)
void lstm_kernel(const unsigned short* __restrict__ cat_bf,  // [B][KIN]
                 const unsigned short* __restrict__ h_bf_rd, // [B][HID]
                 const unsigned short* __restrict__ wih,     // packed layer slice
                 const unsigned short* __restrict__ whh,     // packed layer slice
                 const float* __restrict__ bias,             // [GATES]
                 float* __restrict__ h_f32,                  // [B][HID]
                 float* __restrict__ c_f32,                  // [B][HID]
                 unsigned short* __restrict__ h_bf_wr,       // [B][HID]
                 float* __restrict__ out_h) {                // [B][HID] or null
  __shared__ unsigned short btile[2][4 * 512];   // 2 stages x 4 gate tiles (8 KB)

  const int tid  = threadIdx.x;
  const int lane = tid & 31;
  const int w    = tid >> 5;
  const int hg   = blockIdx.x;
  const int m0   = 16 * w;
  const int hi8  = (lane >> 4) * 8;
  const int row  = m0 + (lane & 15);

  // ---- TDM issue: one k-tile's 4 gate tiles -> LDS stage ----
  auto issue_tdm = [&](int kt, int stage) {
    const bool ih = kt < NKT_IH;
    const unsigned short* src =
        ih ? (wih + (size_t)((hg * 4) * NKT_IH + kt) * 512)
           : (whh + (size_t)((hg * 4) * NKT_HH + (kt - NKT_IH)) * 512);
    const unsigned rstride = ih ? (unsigned)(NKT_IH * 512)   // gate stride (elems)
                                : (unsigned)(NKT_HH * 512);
    unsigned long long ga = (unsigned long long)(uintptr_t)src;
    unsigned lds = (unsigned)(uintptr_t)&btile[stage][0];    // flat addr low32 = LDS offset
    u32x4 g0;
    g0.x = 1u;                                   // count=1 (valid user descriptor)
    g0.y = lds;                                  // D#.lds_addr
    g0.z = (unsigned)ga;                         // D#.global_addr[31:0]
    g0.w = (unsigned)(ga >> 32) | (2u << 30);    // addr[56:32] | type=2 (image)
    i32x8 g1 = {};
    g1[0] = (int)(1u << 16);                     // data_size = 1 -> 2 bytes
    g1[1] = (int)(512u << 16);                   // tensor_dim0 = 512
    g1[2] = (int)(4u << 16);                     // tensor_dim1 = 4
    g1[3] = (int)(512u << 16);                   // tile_dim0 = 512
    g1[4] = 4;                                   // tile_dim1 = 4
    g1[5] = (int)rstride;                        // tensor_dim0_stride
    i32x4 g2 = {}, g3 = {};
    i32x8 g4 = {};                               // unused trailing group (6-arg builtin)
    __builtin_amdgcn_tensor_load_to_lds(g0, g1, g2, g3, g4, 0);
  };

  // ---- A-tile fetch (global, per-wave rows) ----
  auto load_a = [&](int kt, TileBF& a) {
    const uint4v* pa =
        (kt < NKT_IH)
            ? (const uint4v*)(cat_bf + (size_t)row * KIN + kt * 32 + hi8)
            : (const uint4v*)(h_bf_rd + (size_t)row * HID + (kt - NKT_IH) * 32 + hi8);
    __builtin_prefetch(pa + 8, 0, 3);            // same row, 2 k-tiles ahead, WGP-resident hint
    a.u[0] = pa[0];                              // K .. K+7
    a.u[1] = pa[2];                              // K+16 .. K+23
  };

  v8f acc0 = {}, acc1 = {}, acc2 = {}, acc3 = {};

  // prologue: stage 0 in flight, first A tile in regs
  if (w == 0) {
    issue_tdm(0, 0);
    __builtin_amdgcn_s_wait_tensorcnt(0);
  }
  TileBF a_cur, a_nxt;
  load_a(0, a_cur);
  __syncthreads();                               // stage 0 visible to all waves

  for (int kt = 0; kt < NKT_ALL; ++kt) {
    const int cur = kt & 1;
    // after the barrier everyone is done reading stage cur^1 -> safe to refill
    if (w == 0 && kt + 1 < NKT_ALL) issue_tdm(kt + 1, cur ^ 1);
    if (kt + 1 < NKT_ALL) load_a(kt + 1, a_nxt);

    const unsigned short* bp = &btile[cur][0] + lane * 16;
    v16bf b0 = *(const v16bf*)(bp);
    v16bf b1 = *(const v16bf*)(bp + 512);
    v16bf b2 = *(const v16bf*)(bp + 1024);
    v16bf b3 = *(const v16bf*)(bp + 1536);
    acc0 = wmma_bf16(a_cur.v, b0, acc0);
    acc1 = wmma_bf16(a_cur.v, b1, acc1);
    acc2 = wmma_bf16(a_cur.v, b2, acc2);
    acc3 = wmma_bf16(a_cur.v, b3, acc3);

    if (kt + 1 < NKT_ALL) {
      a_cur = a_nxt;
      if (w == 0) __builtin_amdgcn_s_wait_tensorcnt(0);  // next stage landed
      __syncthreads();                                   // publish to all waves
    }
  }

  // ---- epilogue: all 4 gates for (b, hidx) live in the same lane ----
  const int hidx = hg * 16 + (lane & 15);
  const float bi  = bias[hidx];
  const float bf_ = bias[HID + hidx];
  const float bg  = bias[2 * HID + hidx];
  const float bo  = bias[3 * HID + hidx];
  #pragma unroll
  for (int r = 0; r < 8; ++r) {
    const int b   = m0 + r + hi8;
    const int idx = b * HID + hidx;
    float gi = acc0[r] + bi;
    float gf = acc1[r] + bf_;
    float gg = acc2[r] + bg;
    float go = acc3[r] + bo;
    float cn = sigf(gf) * c_f32[idx] + sigf(gi) * tanhf(gg);
    float hn = sigf(go) * tanhf(cn);
    c_f32[idx]   = cn;
    h_f32[idx]   = hn;
    h_bf_wr[idx] = f2bf(hn);
    if (out_h) out_h[idx] = hn;
  }
}

__global__ void final_copy_kernel(const float* __restrict__ h,
                                  const float* __restrict__ c,
                                  float* __restrict__ ht,
                                  float* __restrict__ ct, int n) {
  int i = blockIdx.x * blockDim.x + threadIdx.x;
  if (i < n) { ht[i] = h[i]; ct[i] = c[i]; }
}

// ==================================================================
// Host side
// ==================================================================
extern "C" void kernel_launch(void* const* d_in, const int* in_sizes, int n_in,
                              void* d_out, int out_size, void* d_ws, size_t ws_size,
                              hipStream_t stream) {
  const float* inp = (const float*)d_in[0];
  const float* ctx = (const float*)d_in[1];
  const float* Wih = (const float*)d_in[2];
  const float* Whh = (const float*)d_in[3];
  const float* bih = (const float*)d_in[4];
  const float* bhh = (const float*)d_in[5];
  float* out = (float*)d_out;
  char*  ws  = (char*)d_ws;

  const size_t OFF_CTXB = 0;
  const size_t SZ_CTXB  = (size_t)BATCH * SEQ * DCTX * 2;
  const size_t OFF_WIH  = OFF_CTXB + SZ_CTXB;
  const size_t SZ_WIH   = (size_t)NLAYERS * GATES * KIN * 2;
  const size_t OFF_WHH  = OFF_WIH + SZ_WIH;
  const size_t SZ_WHH   = (size_t)NLAYERS * GATES * HID * 2;
  const size_t OFF_BIAS = OFF_WHH + SZ_WHH;
  const size_t SZ_BIAS  = (size_t)NLAYERS * GATES * 4;
  const size_t OFF_H    = OFF_BIAS + SZ_BIAS;
  const size_t SZ_HF    = (size_t)NLAYERS * BATCH * HID * 4;
  const size_t OFF_C    = OFF_H + SZ_HF;
  const size_t OFF_HBF  = OFF_C + SZ_HF;
  const size_t SZ_HBF   = (size_t)2 * NLAYERS * BATCH * HID * 2;
  const size_t OFF_CAT  = OFF_HBF + SZ_HBF;

  unsigned short* ctxb  = (unsigned short*)(ws + OFF_CTXB);
  unsigned short* wihp  = (unsigned short*)(ws + OFF_WIH);
  unsigned short* whhp  = (unsigned short*)(ws + OFF_WHH);
  float*          biasS = (float*)(ws + OFF_BIAS);
  float*          hstat = (float*)(ws + OFF_H);
  float*          cstat = (float*)(ws + OFF_C);
  unsigned short* hbf   = (unsigned short*)(ws + OFF_HBF);
  unsigned short* catb  = (unsigned short*)(ws + OFF_CAT);

  const size_t O_OUT = 0;
  const size_t O_CTX = (size_t)T_STEPS * BATCH * HID;
  const size_t O_ATT = O_CTX + (size_t)T_STEPS * BATCH * DCTX;
  const size_t O_HT  = O_ATT + (size_t)T_STEPS * BATCH * SEQ;
  const size_t O_CT  = O_HT + (size_t)NLAYERS * BATCH * HID;

  {
    int n = NLAYERS * GATES * KIN;
    pack_wih_kernel<<<(n + 255) / 256, 256, 0, stream>>>(Wih, wihp);
  }
  {
    int n = NLAYERS * GATES * HID;
    pack_whh_kernel<<<(n + 255) / 256, 256, 0, stream>>>(Whh, whhp);
  }
  {
    int n = NLAYERS * GATES;
    bias_sum_kernel<<<(n + 255) / 256, 256, 0, stream>>>(bih, bhh, biasS, n);
  }
  {
    int n2 = BATCH * SEQ * DCTX / 2;
    ctx2bf_kernel<<<(n2 + 255) / 256, 256, 0, stream>>>(ctx, ctxb, n2);
  }
  (void)hipMemsetAsync(ws + OFF_H, 0, 2 * SZ_HF + SZ_HBF, stream);

  const size_t HS = (size_t)BATCH * HID;

  for (int t = 0; t < T_STEPS; ++t) {
    const int rp = t & 1, wp = (t + 1) & 1;

    attn_kernel<<<BATCH, 256, 0, stream>>>(
        inp + (size_t)t * BATCH * HID, ctxb, catb, nullptr, nullptr);
    lstm_kernel<<<32, 128, 0, stream>>>(
        catb, hbf + ((size_t)rp * NLAYERS + 0) * HS,
        wihp, whhp, biasS,
        hstat, cstat,
        hbf + ((size_t)wp * NLAYERS + 0) * HS,
        nullptr);

    attn_kernel<<<BATCH, 256, 0, stream>>>(
        hstat, ctxb, catb,
        out + O_ATT + (size_t)t * BATCH * SEQ,
        out + O_CTX + (size_t)t * BATCH * DCTX);
    lstm_kernel<<<32, 128, 0, stream>>>(
        catb, hbf + ((size_t)rp * NLAYERS + 1) * HS,
        wihp + (size_t)GATES * KIN, whhp + (size_t)GATES * HID, biasS + GATES,
        hstat + HS, cstat + HS,
        hbf + ((size_t)wp * NLAYERS + 1) * HS,
        out + O_OUT + (size_t)t * BATCH * HID);
  }

  {
    int n = NLAYERS * BATCH * HID;
    final_copy_kernel<<<(n + 255) / 256, 256, 0, stream>>>(
        hstat, cstat, out + O_HT, out + O_CT, n);
  }
}
